// SwinTransformerBlock_52639119180082
// MI455X (gfx1250) — matrix operand order repaired and checked by reading, not compiled, not run on hardware
//
#include <hip/hip_runtime.h>
#include <math.h>

typedef __bf16 bf16_t;
typedef __attribute__((ext_vector_type(16))) __bf16 v16bf;
typedef __attribute__((ext_vector_type(8)))  float  v8f;
typedef __attribute__((ext_vector_type(4)))  int    v4i;

#define WMMA_BF16(a, b, c) \
  __builtin_amdgcn_wmma_f32_16x16x32_bf16(false, (a), false, (b), (short)0, (c), false, false)

// --------- gfx1250 async global->LDS copy (ASYNCcnt path), with fallback ----
#if defined(__HIP_DEVICE_COMPILE__) && \
    __has_builtin(__builtin_amdgcn_global_load_async_to_lds_b128) && \
    __has_builtin(__builtin_amdgcn_s_wait_asynccnt)
#define HAS_ASYNC_LDS 1
#else
#define HAS_ASYNC_LDS 0
#endif

__device__ inline void async_cp16(void* lds_dst, const void* gsrc) {
#if HAS_ASYNC_LDS
  typedef __attribute__((address_space(1))) v4i* gptr_t;   // global int4*
  typedef __attribute__((address_space(3))) v4i* lptr_t;   // LDS int4* (32-bit)
  __builtin_amdgcn_global_load_async_to_lds_b128(
      (gptr_t)(uintptr_t)gsrc,
      (lptr_t)(unsigned int)(uintptr_t)lds_dst, 0, 0);
#else
  *(uint4*)lds_dst = *(const uint4*)gsrc;
#endif
}
__device__ inline void async_join() {
#if HAS_ASYNC_LDS
  __builtin_amdgcn_s_wait_asynccnt(0);   // s_wait_asynccnt 0x0
#endif
}

// ---------------- problem constants ----------------
// B=16, H=W=64, C=512, heads=16, head_dim=32, window 8x8 (N=64), shift 4
static constexpr int  C_DIM   = 512;
static constexpr int  HEADS   = 16;
static constexpr int  HD      = 32;
static constexpr int  NTOK    = 64;     // tokens per window
static constexpr int  BWIN    = 1024;   // total windows
static constexpr int  MROWS   = 65536;  // BWIN * NTOK
static constexpr int  DFF     = 2048;
static constexpr long QKV_T_STRIDE = (long)BWIN * HEADS * NTOK * HD; // elems per q/k/v tensor

// ============================================================
// K0: weights -> bf16, concat qkv bias
// ============================================================
__global__ __launch_bounds__(256) void prep_kernel(
    const float* __restrict__ qkv_w, const float* __restrict__ proj_w,
    const float* __restrict__ fc1_w, const float* __restrict__ fc2_w,
    const float* __restrict__ q_bias, const float* __restrict__ v_bias,
    bf16_t* __restrict__ wqkv, bf16_t* __restrict__ wproj,
    bf16_t* __restrict__ wfc1, bf16_t* __restrict__ wfc2,
    float* __restrict__ qkvb) {
  int i = blockIdx.x * 256 + threadIdx.x;
  if (i < 3 * C_DIM * C_DIM) wqkv[i] = (bf16_t)qkv_w[i];
  if (i < C_DIM * C_DIM)     wproj[i] = (bf16_t)proj_w[i];
  if (i < DFF * C_DIM)       { wfc1[i] = (bf16_t)fc1_w[i]; wfc2[i] = (bf16_t)fc2_w[i]; }
  if (i < 3 * C_DIM)
    qkvb[i] = (i < C_DIM) ? q_bias[i] : (i < 2 * C_DIM ? 0.0f : v_bias[i - 2 * C_DIM]);
}

// ============================================================
// K1: continuous position bias MLP: 225 grid points -> 16 heads
// ============================================================
__device__ inline float cpb_coord(int i) {
  float t = (float)(i - 7) * (8.0f / 7.0f);            // /(PWS-1)*8
  float a = log2f(fabsf(t) + 1.0f) * (1.0f / log2f(8.0f));
  return (t > 0.0f) ? a : ((t < 0.0f) ? -a : 0.0f);
}

__global__ __launch_bounds__(256) void cpb_hidden_kernel(
    const float* __restrict__ w1, const float* __restrict__ b1,
    const float* __restrict__ w2, float* __restrict__ bias_table /*225x16*/) {
  int p = blockIdx.x;                 // 0..224
  float ti = cpb_coord(p / 15);
  float tj = cpb_coord(p % 15);
  __shared__ float hid[512];
  for (int u = threadIdx.x; u < 512; u += 256)
    hid[u] = fmaxf(ti * w1[u * 2 + 0] + tj * w1[u * 2 + 1] + b1[u], 0.0f);
  __syncthreads();
  if (threadIdx.x < HEADS) {
    int hh = threadIdx.x;
    float s = 0.0f;
    for (int u = 0; u < 512; ++u) s += hid[u] * w2[hh * 512 + u];
    bias_table[p * HEADS + hh] = s;
  }
}

__global__ __launch_bounds__(256) void cpb_rpb_kernel(
    const float* __restrict__ bias_table, float* __restrict__ rpb /*16x64x64*/) {
  int t = blockIdx.x * 256 + threadIdx.x;   // 4096 (n,m) pairs
  int n = t >> 6, m = t & 63;
  int yn = n >> 3, xn = n & 7, ym = m >> 3, xm = m & 7;
  int idx = (yn - ym + 7) * 15 + (xn - xm + 7);
#pragma unroll
  for (int hh = 0; hh < HEADS; ++hh) {
    float v = bias_table[idx * HEADS + hh];
    rpb[((size_t)hh * 64 + n) * 64 + m] = 16.0f / (1.0f + __expf(-v));
  }
}

// ============================================================
// K2: roll(-4,-4) + window partition + fp32 -> bf16
// ============================================================
__global__ __launch_bounds__(256) void make_windows_kernel(
    const float* __restrict__ x, bf16_t* __restrict__ xw) {
  size_t t = (size_t)blockIdx.x * 256 + threadIdx.x;   // 65536*128 float4 chunks
  size_t r = t >> 7;
  int c4 = (int)(t & 127);
  int b_ = (int)(r >> 6), n = (int)(r & 63);
  int b = b_ >> 6, wi = b_ & 63;
  int wy = wi >> 3, wx = wi & 7, iy = n >> 3, ix = n & 7;
  int h = ((wy * 8 + iy) + 4) & 63;                    // shifted[hs] = x[(hs+4)%64]
  int w = ((wx * 8 + ix) + 4) & 63;
  const float4 v = *(const float4*)(x + (((size_t)b * 4096) + h * 64 + w) * C_DIM + c4 * 4);
  bf16_t* dst = xw + r * C_DIM + c4 * 4;
  dst[0] = (bf16_t)v.x; dst[1] = (bf16_t)v.y; dst[2] = (bf16_t)v.z; dst[3] = (bf16_t)v.w;
}

// ============================================================
// Generic WMMA bf16 GEMM: C[m][n] = sum_k A[m][k]*B[n][k] + bias[n]
// 128x128 tile / block (8 waves, each 32x64 = 2x4 accs), BK=32,
// double-buffered LDS fed by GLOBAL_LOAD_ASYNC_TO_LDS_B128.
// MODE 0: fp32 store. MODE 1: GELU -> bf16. MODE 2: QKV scatter -> bf16.
// ============================================================
template <int MODE>
__global__ __launch_bounds__(256) void gemm_bf16_kernel(
    const bf16_t* __restrict__ A, const bf16_t* __restrict__ Bm,
    const float* __restrict__ bias, void* __restrict__ Cout,
    int N, int K) {
  __shared__ alignas(16) bf16_t sA[2][128 * 32];
  __shared__ alignas(16) bf16_t sB[2][128 * 32];
  const int tid = threadIdx.x;
  const int wave = tid >> 5, lane = tid & 31;
  const int rowBase = (wave & 3) * 32;     // 4 waves along M
  const int colBase = (wave >> 2) * 64;    // 2 waves along N
  const int m0 = blockIdx.y * 128;
  const int n0 = blockIdx.x * 128;

  auto load_stage = [&](int buf, int k0) {
#pragma unroll
    for (int i = 0; i < 2; ++i) {          // 512 16B chunks per tile
      int qd = tid * 2 + i;
      int r = qd >> 2, cc = qd & 3;
      async_cp16(&sA[buf][r * 32 + cc * 8], A + (size_t)(m0 + r) * K + k0 + cc * 8);
      async_cp16(&sB[buf][r * 32 + cc * 8], Bm + (size_t)(n0 + r) * K + k0 + cc * 8);
    }
  };

  v8f acc[2][4];
#pragma unroll
  for (int mt = 0; mt < 2; ++mt)
#pragma unroll
    for (int nt = 0; nt < 4; ++nt) acc[mt][nt] = (v8f){0,0,0,0,0,0,0,0};

  const int nk = K >> 5;
  load_stage(0, 0);
  async_join();
  __syncthreads();

  int buf = 0;
  for (int kt = 0; kt < nk; ++kt) {
    if (kt + 1 < nk) load_stage(buf ^ 1, (kt + 1) * 32);   // prefetch next tile

    v16bf af[2], bf[4];
    const int mrow = lane & 15;
    const int kca = (lane & 16) ? 8 : 0;     // A lane K-chunk split
#pragma unroll
    for (int mt = 0; mt < 2; ++mt) {
      const bf16_t* p = &sA[buf][(rowBase + mt * 16 + mrow) * 32];
#pragma unroll
      for (int i = 0; i < 8; ++i) { af[mt][i] = p[kca + i]; af[mt][8 + i] = p[kca + 16 + i]; }
    }
    const int kcb = (lane & 16) ? 8 : 0;     // B lane K-chunk split
#pragma unroll
    for (int nt = 0; nt < 4; ++nt) {
      const bf16_t* p = &sB[buf][(colBase + nt * 16 + (lane & 15)) * 32];
#pragma unroll
      for (int i = 0; i < 8; ++i) { bf[nt][i] = p[kcb + i]; bf[nt][8 + i] = p[kcb + 16 + i]; }
    }
#pragma unroll
    for (int mt = 0; mt < 2; ++mt)
#pragma unroll
      for (int nt = 0; nt < 4; ++nt) acc[mt][nt] = WMMA_BF16(af[mt], bf[nt], acc[mt][nt]);

    async_join();
    __syncthreads();
    buf ^= 1;
  }

  // epilogue: lane l, vgpr j -> M = base + j + (l&16 ? 8 : 0), N = base + (l&15)
#pragma unroll
  for (int mt = 0; mt < 2; ++mt)
#pragma unroll
    for (int nt = 0; nt < 4; ++nt)
#pragma unroll
      for (int j = 0; j < 8; ++j) {
        int m = m0 + rowBase + mt * 16 + j + ((lane & 16) ? 8 : 0);
        int n = n0 + colBase + nt * 16 + (lane & 15);
        float v = acc[mt][nt][j] + bias[n];
        if (MODE == 0) {
          ((float*)Cout)[(size_t)m * N + n] = v;
        } else if (MODE == 1) {
          float g = 0.5f * v * (1.0f + erff(v * 0.70710678f));
          ((bf16_t*)Cout)[(size_t)m * N + n] = (bf16_t)g;
        } else {                              // QKV scatter: n in [0,1536)
          int tsel = n >> 9;                  // 0=q 1=k 2=v
          int head = (n & 511) >> 5;
          int d    = n & 31;
          int b_   = m >> 6, nn = m & 63;
          ((bf16_t*)Cout)[(size_t)tsel * QKV_T_STRIDE +
                          (((size_t)b_ * HEADS + head) * NTOK + nn) * HD + d] = (bf16_t)v;
        }
      }
}

// ============================================================
// K4: windowed cosine attention, one block (4 waves) per (window, head)
// ============================================================
__global__ __launch_bounds__(128) void attn_kernel(
    const bf16_t* __restrict__ q, const bf16_t* __restrict__ k,
    const bf16_t* __restrict__ v, const float* __restrict__ rpb,
    const float* __restrict__ mask, const float* __restrict__ logit_scale,
    bf16_t* __restrict__ outw) {
  const int bh = blockIdx.x;
  const int b_ = bh >> 4, h = bh & 15;
  __shared__ alignas(16) bf16_t sq[64 * 32];
  __shared__ alignas(16) bf16_t sk[64 * 32];
  __shared__ alignas(16) bf16_t svT[32 * 64];
  __shared__ alignas(16) float  S[64 * 64];
  __shared__ alignas(16) bf16_t P[64 * 64];
  const int tid = threadIdx.x;
  const size_t base = ((size_t)(b_ * HEADS + h)) * NTOK * HD;

  // q, k: straight contiguous copies -> async-to-LDS path
#pragma unroll
  for (int c = tid; c < 256; c += 128) {     // 256 16B chunks each
    async_cp16(sq + c * 8, q + base + c * 8);
    async_cp16(sk + c * 8, k + base + c * 8);
  }
  // v: transposed on the fly (sync path)
  for (int i = tid; i < NTOK * HD; i += 128) {
    int m = i >> 5, d = i & 31;
    svT[d * 64 + m] = v[base + i];
  }
  async_join();
  __syncthreads();

  const float scale = __expf(fminf(logit_scale[h], 4.60517019f)); // ln(100)
  if (tid < 64) {                      // normalize q row, fold in scale
    float s = 0.0f;
#pragma unroll
    for (int d = 0; d < 32; ++d) { float x = (float)sq[tid * 32 + d]; s += x * x; }
    float inv = scale / fmaxf(sqrtf(s), 1e-12f);
#pragma unroll
    for (int d = 0; d < 32; ++d) sq[tid * 32 + d] = (bf16_t)((float)sq[tid * 32 + d] * inv);
  } else {                             // normalize k row
    int r = tid - 64;
    float s = 0.0f;
#pragma unroll
    for (int d = 0; d < 32; ++d) { float x = (float)sk[r * 32 + d]; s += x * x; }
    float inv = 1.0f / fmaxf(sqrtf(s), 1e-12f);
#pragma unroll
    for (int d = 0; d < 32; ++d) sk[r * 32 + d] = (bf16_t)((float)sk[r * 32 + d] * inv);
  }
  __syncthreads();

  const int wave = tid >> 5, lane = tid & 31;
  const int rb = wave * 16;
  const float* rpbh  = rpb  + (size_t)h * 64 * 64;
  const float* maskw = mask + (size_t)(b_ & 63) * 64 * 64;

  // S = qn @ kn^T  (M=64 split over 4 waves, N=64, K=32)
  v16bf aq;
  {
    const int kca = (lane & 16) ? 8 : 0;
    const bf16_t* p = sq + (rb + (lane & 15)) * 32;
#pragma unroll
    for (int i = 0; i < 8; ++i) { aq[i] = p[kca + i]; aq[8 + i] = p[kca + 16 + i]; }
  }
#pragma unroll
  for (int nt = 0; nt < 4; ++nt) {
    v16bf bk;
    const int kcb = (lane & 16) ? 8 : 0;
    const bf16_t* p = sk + (nt * 16 + (lane & 15)) * 32;
#pragma unroll
    for (int i = 0; i < 8; ++i) { bk[i] = p[kcb + i]; bk[8 + i] = p[kcb + 16 + i]; }
    v8f a = (v8f){0,0,0,0,0,0,0,0};
    a = WMMA_BF16(aq, bk, a);
#pragma unroll
    for (int j = 0; j < 8; ++j) {
      int m = rb + j + ((lane & 16) ? 8 : 0);
      int n = nt * 16 + (lane & 15);
      S[m * 64 + n] = a[j] + rpbh[m * 64 + n] + maskw[m * 64 + n];
    }
  }
  __syncthreads();

  if (tid < 64) {                      // fp32 softmax over each row
    float mx = -1e30f;
    for (int n = 0; n < 64; ++n) mx = fmaxf(mx, S[tid * 64 + n]);
    float sm = 0.0f;
    for (int n = 0; n < 64; ++n) sm += __expf(S[tid * 64 + n] - mx);
    float inv = 1.0f / sm;
    for (int n = 0; n < 64; ++n)
      P[tid * 64 + n] = (bf16_t)(__expf(S[tid * 64 + n] - mx) * inv);
  }
  __syncthreads();

  // out = P @ v  (M=64/4 waves, N=32, K=64)
  v8f oacc[2] = {(v8f){0,0,0,0,0,0,0,0}, (v8f){0,0,0,0,0,0,0,0}};
#pragma unroll
  for (int ks = 0; ks < 2; ++ks) {
    v16bf ap;
    const int kca = (lane & 16) ? 8 : 0;
    const bf16_t* pa = P + (rb + (lane & 15)) * 64 + ks * 32;
#pragma unroll
    for (int i = 0; i < 8; ++i) { ap[i] = pa[kca + i]; ap[8 + i] = pa[kca + 16 + i]; }
#pragma unroll
    for (int nt = 0; nt < 2; ++nt) {
      v16bf bv;
      const int kcb = (lane & 16) ? 8 : 0;
      const bf16_t* pb = svT + (nt * 16 + (lane & 15)) * 64 + ks * 32;
#pragma unroll
      for (int i = 0; i < 8; ++i) { bv[i] = pb[kcb + i]; bv[8 + i] = pb[kcb + 16 + i]; }
      oacc[nt] = WMMA_BF16(ap, bv, oacc[nt]);
    }
  }
#pragma unroll
  for (int nt = 0; nt < 2; ++nt)
#pragma unroll
    for (int j = 0; j < 8; ++j) {
      int m = rb + j + ((lane & 16) ? 8 : 0);
      int d = nt * 16 + (lane & 15);
      outw[((size_t)b_ * NTOK + m) * C_DIM + h * HD + d] = (bf16_t)oacc[nt][j];
    }
}

// ============================================================
// K6/K9: (optional window-reverse permute) + LayerNorm + residual
// permute=1: src rows are in window layout, residual = original x
// ============================================================
__global__ __launch_bounds__(128) void ln_residual_kernel(
    const float* __restrict__ src, const float* __restrict__ resid,
    const float* __restrict__ g, const float* __restrict__ bv,
    float* __restrict__ outf, bf16_t* __restrict__ outb, int permute) {
  const int tok = blockIdx.x;          // 0..65535
  size_t srow;
  if (permute) {
    int b = tok >> 12, l = tok & 4095;
    int h = l >> 6, w = l & 63;
    int hr = (h + 60) & 63, wr = (w + 60) & 63;    // inverse roll(+4,+4)
    int wy = hr >> 3, iy = hr & 7, wx = wr >> 3, ix = wr & 7;
    srow = (((size_t)(b * 64 + wy * 8 + wx)) * 64 + (iy * 8 + ix)) * C_DIM;
  } else {
    srow = (size_t)tok * C_DIM;
  }
  __shared__ float red[128];
  const int tid = threadIdx.x;
  float4 v4 = *(const float4*)(src + srow + tid * 4);
  float vals[4] = {v4.x, v4.y, v4.z, v4.w};
  float s = vals[0] + vals[1] + vals[2] + vals[3];
  red[tid] = s; __syncthreads();
#pragma unroll
  for (int off = 64; off > 0; off >>= 1) {
    if (tid < off) red[tid] += red[tid + off];
    __syncthreads();
  }
  float mean = red[0] * (1.0f / 512.0f);
  __syncthreads();
  float s2 = 0.0f;
#pragma unroll
  for (int i = 0; i < 4; ++i) { float d = vals[i] - mean; s2 += d * d; }
  red[tid] = s2; __syncthreads();
#pragma unroll
  for (int off = 64; off > 0; off >>= 1) {
    if (tid < off) red[tid] += red[tid + off];
    __syncthreads();
  }
  float rstd = rsqrtf(red[0] * (1.0f / 512.0f) + 1e-5f);
  const size_t orow = (size_t)tok * C_DIM;
  float4 r4 = *(const float4*)(resid + orow + tid * 4);
  float rs[4] = {r4.x, r4.y, r4.z, r4.w};
#pragma unroll
  for (int i = 0; i < 4; ++i) {
    int c = tid * 4 + i;
    float o = rs[i] + (vals[i] - mean) * rstd * g[c] + bv[c];
    outf[orow + c] = o;
    if (outb) outb[orow + c] = (bf16_t)o;
  }
}

// ============================================================
// host side
// ============================================================
extern "C" void kernel_launch(void* const* d_in, const int* in_sizes, int n_in,
                              void* d_out, int out_size, void* d_ws, size_t ws_size,
                              hipStream_t stream) {
  (void)in_sizes; (void)n_in; (void)out_size; (void)ws_size;
  const float* x        = (const float*)d_in[0];
  const float* qkv_w    = (const float*)d_in[1];
  const float* q_bias   = (const float*)d_in[2];
  const float* v_bias   = (const float*)d_in[3];
  const float* proj_w   = (const float*)d_in[4];
  const float* proj_b   = (const float*)d_in[5];
  const float* lscale   = (const float*)d_in[6];
  const float* cpb_w1   = (const float*)d_in[7];
  const float* cpb_b1   = (const float*)d_in[8];
  const float* cpb_w2   = (const float*)d_in[9];
  const float* norm1_g  = (const float*)d_in[10];
  const float* norm1_b  = (const float*)d_in[11];
  const float* norm2_g  = (const float*)d_in[12];
  const float* norm2_b  = (const float*)d_in[13];
  const float* fc1_w    = (const float*)d_in[14];
  const float* fc1_b    = (const float*)d_in[15];
  const float* fc2_w    = (const float*)d_in[16];
  const float* fc2_b    = (const float*)d_in[17];
  const float* mask     = (const float*)d_in[20];

  char* ws = (char*)d_ws;
  // --- workspace layout (bytes) ---
  const size_t OFF_WQKV = 0;                             // 1536*512 bf16
  const size_t OFF_WPROJ = OFF_WQKV + 1572864;           // 512*512 bf16
  const size_t OFF_WFC1 = OFF_WPROJ + 524288;            // 2048*512 bf16
  const size_t OFF_WFC2 = OFF_WFC1 + 2097152;            // 512*2048 bf16
  const size_t OFF_QKVB = OFF_WFC2 + 2097152;            // 1536 f32
  const size_t OFF_BT   = OFF_QKVB + 8192;               // 225*16 f32
  const size_t OFF_RPB  = OFF_BT + 16384;                // 16*64*64 f32
  const size_t OFF_R1 = 8u  * 1024 * 1024;               // 64MB  xw / attn-out bf16
  const size_t OFF_R2 = 72u * 1024 * 1024;               // 256MB qkv -> hdd bf16
  const size_t OFF_R3 = 328u * 1024 * 1024;              // 128MB x1 fp32
  const size_t OFF_R4 = 456u * 1024 * 1024;              // 64MB  x1 bf16
  const size_t OFF_R5 = 520u * 1024 * 1024;              // 128MB projout / mlpout fp32

  bf16_t* wqkv = (bf16_t*)(ws + OFF_WQKV);
  bf16_t* wproj = (bf16_t*)(ws + OFF_WPROJ);
  bf16_t* wfc1 = (bf16_t*)(ws + OFF_WFC1);
  bf16_t* wfc2 = (bf16_t*)(ws + OFF_WFC2);
  float*  qkvb = (float*)(ws + OFF_QKVB);
  float*  btab = (float*)(ws + OFF_BT);
  float*  rpb  = (float*)(ws + OFF_RPB);
  bf16_t* R1   = (bf16_t*)(ws + OFF_R1);
  bf16_t* R2   = (bf16_t*)(ws + OFF_R2);
  float*  R3   = (float*)(ws + OFF_R3);
  bf16_t* R4   = (bf16_t*)(ws + OFF_R4);
  float*  R5   = (float*)(ws + OFF_R5);

  // K0: weight conversion
  prep_kernel<<<3072, 256, 0, stream>>>(qkv_w, proj_w, fc1_w, fc2_w, q_bias, v_bias,
                                        wqkv, wproj, wfc1, wfc2, qkvb);
  // K1: relative position bias
  cpb_hidden_kernel<<<225, 256, 0, stream>>>(cpb_w1, cpb_b1, cpb_w2, btab);
  cpb_rpb_kernel<<<16, 256, 0, stream>>>(btab, rpb);
  // K2: shift + window partition + bf16
  make_windows_kernel<<<32768, 256, 0, stream>>>(x, R1);
  // K3: QKV GEMM (65536 x 1536 x 512), scatter into [b_,h,n,d]
  gemm_bf16_kernel<2><<<dim3(12, 512), 256, 0, stream>>>(R1, wqkv, qkvb, (void*)R2,
                                                         3 * C_DIM, C_DIM);
  // K4: windowed attention
  attn_kernel<<<BWIN * HEADS, 128, 0, stream>>>(
      R2, R2 + QKV_T_STRIDE, R2 + 2 * QKV_T_STRIDE, rpb, mask, lscale, R1);
  // K5: proj GEMM (65536 x 512 x 512) -> fp32
  gemm_bf16_kernel<0><<<dim3(4, 512), 256, 0, stream>>>(R1, wproj, proj_b, (void*)R5,
                                                        C_DIM, C_DIM);
  // K6: window reverse + LN + residual -> x1 (fp32 + bf16)
  ln_residual_kernel<<<MROWS, 128, 0, stream>>>(R5, x, norm1_g, norm1_b, R3, R4, 1);
  // K7: fc1 GEMM (65536 x 2048 x 512) + GELU -> bf16
  gemm_bf16_kernel<1><<<dim3(16, 512), 256, 0, stream>>>(R4, wfc1, fc1_b, (void*)R2,
                                                         DFF, C_DIM);
  // K8: fc2 GEMM (65536 x 512 x 2048) -> fp32
  gemm_bf16_kernel<0><<<dim3(4, 512), 256, 0, stream>>>(R2, wfc2, fc2_b, (void*)R5,
                                                        C_DIM, DFF);
  // K9: LN + residual -> d_out
  ln_residual_kernel<<<MROWS, 128, 0, stream>>>(R5, R3, norm2_g, norm2_b,
                                                (float*)d_out, nullptr, 0);
}
